// EnhancedGNNDetector_26018911879665
// MI455X (gfx1250) — matrix operand order repaired and checked by reading, not compiled, not run on hardware
//
#include <hip/hip_runtime.h>
#include <hip/hip_bf16.h>
#include <math.h>

// ---------------------------------------------------------------- types ----
typedef __attribute__((ext_vector_type(16))) __bf16 v16bf;
typedef __attribute__((ext_vector_type(8)))  __bf16 v8bf;
typedef __attribute__((ext_vector_type(8)))  float  v8f;

#define D_IN 128
#define HID  256
#define FH   64
#define HEADS 4

// ------------------------------------------------------------- utilities ---
__global__ void fill_f32(float* __restrict__ p, float v, long long n) {
    long long i = (long long)blockIdx.x * blockDim.x + threadIdx.x;
    long long stride = (long long)gridDim.x * blockDim.x;
    for (; i < n; i += stride) p[i] = v;
}

__global__ void convert_f32_to_bf16(const float* __restrict__ src,
                                    __bf16* __restrict__ dst, long long n) {
    long long i = (long long)blockIdx.x * blockDim.x + threadIdx.x;
    long long stride = (long long)gridDim.x * blockDim.x;
    for (; i < n; i += stride) dst[i] = (__bf16)src[i];
}

// Pre-swizzle weight W[K,N] (row-major f32) into WMMA B-fragment order:
// Bsw[((ct*KT + kt)*32 + lane)*16 + i] = bf16( W[(kt*32 + (i>>3)*16 + (lane>>4)*8 + (i&7))*N + ct*16 + (lane&15)] )
__global__ void swizzle_weight(const float* __restrict__ W, __bf16* __restrict__ Bsw,
                               int K, int N) {
    int idx = blockIdx.x * blockDim.x + threadIdx.x;
    int KT = K / 32;
    int total = (N / 16) * KT * 512;
    if (idx >= total) return;
    int i    = idx & 15;
    int lane = (idx >> 4) & 31;
    int kt   = (idx >> 9) % KT;
    int ct   = (idx >> 9) / KT;
    int k = kt * 32 + (i >> 3) * 16 + (lane >> 4) * 8 + (i & 7);
    int n = ct * 16 + (lane & 15);
    Bsw[idx] = (__bf16)W[(size_t)k * N + n];
}

// ------------------------------------------------------------ WMMA GEMM ----
// C[M,N] = A[M,K] @ W[K,N]   A: bf16 row-major, W: pre-swizzled bf16, C: f32.
// One wave per 16x16 tile of C; block covers 16 rows x N cols (N/16 waves).
template <int K, int N>
__global__ __launch_bounds__(2 * N)
void gemm_bf16_wmma(const __bf16* __restrict__ A, const __bf16* __restrict__ Bsw,
                    float* __restrict__ C) {
    const int lane = threadIdx.x & 31;
    const int ct   = threadIdx.x >> 5;          // column tile (0 .. N/16-1)
    const int bm   = blockIdx.x * 16;           // row-tile base
    const int r    = bm + (lane & 15);          // A row for this lane
    const int kb   = (lane >> 4) * 8;           // A K sub-base per half-wave
    constexpr int KT = K / 32;

    v8f acc = {};
    #pragma unroll
    for (int kt = 0; kt < KT; ++kt) {
        const int k0 = kt * 32;
        v8bf lo = *(const v8bf*)(A + (size_t)r * K + k0 + kb);
        v8bf hi = *(const v8bf*)(A + (size_t)r * K + k0 + kb + 16);
        v16bf a;
        #pragma unroll
        for (int i = 0; i < 8; ++i) { a[i] = lo[i]; a[8 + i] = hi[i]; }
        v16bf b = *(const v16bf*)(Bsw + (((size_t)ct * KT + kt) * 32 + lane) * 16);
        acc = __builtin_amdgcn_wmma_f32_16x16x32_bf16(
                  false, a, false, b, (short)0, acc, false, false);
    }
    const int col = ct * 16 + (lane & 15);
    const int row0 = bm + (lane >> 4) * 8;      // VGPR j -> row row0+j
    #pragma unroll
    for (int j = 0; j < 8; ++j)
        C[(size_t)(row0 + j) * N + col] = acc[j];
}

// ---------------------------------------------------------- graph kernels --
__global__ void compute_deg(const int* __restrict__ ei, float* __restrict__ deg, int E) {
    int i = blockIdx.x * blockDim.x + threadIdx.x;
    if (i < E) atomicAdd(deg + ei[E + i], 1.0f);
}

__global__ void compute_dinv(const float* __restrict__ deg, float* __restrict__ dinv, int Nn) {
    int i = blockIdx.x * blockDim.x + threadIdx.x;
    if (i < Nn) dinv[i] = rsqrtf(deg[i] + 1.0f);     // +1 = self loop
}

// out[dst,f] += t[src,f] * dinv[src]*dinv[dst];  one block per (edge|selfloop), blockDim=D
__global__ void gcn_aggregate(const float* __restrict__ t, const float* __restrict__ dinv,
                              const int* __restrict__ ei, float* __restrict__ out, int E) {
    int i = blockIdx.x;
    int s, d;
    if (i < E) { s = ei[i]; d = ei[E + i]; } else { s = d = i - E; }
    float norm = dinv[s] * dinv[d];
    int D = blockDim.x;
    int f = threadIdx.x;
    atomicAdd(out + (size_t)d * D + f, t[(size_t)s * D + f] * norm);
}

// v = relu(agg + bias); optional residual add; write optional f32 + bf16 copies
__global__ void gcn_epilogue(const float* __restrict__ agg, const float* __restrict__ bias,
                             const float* __restrict__ residual, float* __restrict__ outf,
                             __bf16* __restrict__ outbf, int D, long long total) {
    long long i = (long long)blockIdx.x * blockDim.x + threadIdx.x;
    long long stride = (long long)gridDim.x * blockDim.x;
    for (; i < total; i += stride) {
        int f = (int)(i % D);
        float v = agg[i] + bias[f];
        v = v > 0.f ? v : 0.f;
        if (residual) v += residual[i];
        if (outf) outf[i] = v;
        outbf[i] = (__bf16)v;
    }
}

// ------------------------------------------------------------- GAT stage ---
__global__ void gat_node_attn(const float* __restrict__ hg, const float* __restrict__ a_src,
                              const float* __restrict__ a_dst, float* __restrict__ als,
                              float* __restrict__ ald, int Nn) {
    int i = blockIdx.x * blockDim.x + threadIdx.x;   // over Nn*HEADS
    if (i >= Nn * HEADS) return;
    int h = i & (HEADS - 1);
    int n = i >> 2;
    const float* base = hg + (size_t)n * HID + h * FH;
    float s = 0.f, d = 0.f;
    #pragma unroll 4
    for (int f = 0; f < FH; ++f) {
        float v = base[f];
        s += v * a_src[h * FH + f];
        d += v * a_dst[h * FH + f];
    }
    als[i] = s;
    ald[i] = d;
}

__device__ inline void atomicMaxFloat(float* addr, float val) {
    if (val >= 0.f) atomicMax((int*)addr, __float_as_int(val));
    else            atomicMin((unsigned int*)addr, __float_as_uint(val));
}

__device__ inline float leaky02(float x) { return x > 0.f ? x : 0.2f * x; }

__global__ void gat_edge_max(const int* __restrict__ ei, const float* __restrict__ als,
                             const float* __restrict__ ald, float* __restrict__ mmax,
                             int E, int EN) {
    int i = blockIdx.x * blockDim.x + threadIdx.x;   // over EN*HEADS
    if (i >= EN * HEADS) return;
    int h = i & (HEADS - 1);
    int idx = i >> 2;
    int s, d;
    if (idx < E) { s = ei[idx]; d = ei[E + idx]; } else { s = d = idx - E; }
    float e = leaky02(als[s * HEADS + h] + ald[d * HEADS + h]);
    atomicMaxFloat(mmax + d * HEADS + h, e);
}

__global__ void gat_edge_den(const int* __restrict__ ei, const float* __restrict__ als,
                             const float* __restrict__ ald, const float* __restrict__ mmax,
                             float* __restrict__ den, int E, int EN) {
    int i = blockIdx.x * blockDim.x + threadIdx.x;
    if (i >= EN * HEADS) return;
    int h = i & (HEADS - 1);
    int idx = i >> 2;
    int s, d;
    if (idx < E) { s = ei[idx]; d = ei[E + idx]; } else { s = d = idx - E; }
    float e = leaky02(als[s * HEADS + h] + ald[d * HEADS + h]);
    atomicAdd(den + d * HEADS + h, __expf(e - mmax[d * HEADS + h]));
}

// out[dst,f] += hg[src,f] * alpha[head(f)];  one block (HID threads) per edge
__global__ void gat_edge_agg(const int* __restrict__ ei, const float* __restrict__ hg,
                             const float* __restrict__ als, const float* __restrict__ ald,
                             const float* __restrict__ mmax, const float* __restrict__ den,
                             float* __restrict__ out, int E) {
    int i = blockIdx.x;
    int s, d;
    if (i < E) { s = ei[i]; d = ei[E + i]; } else { s = d = i - E; }
    int f = threadIdx.x;
    int h = f >> 6;                                  // f / FH
    float e = leaky02(als[s * HEADS + h] + ald[d * HEADS + h]);
    float alpha = __expf(e - mmax[d * HEADS + h]) / den[d * HEADS + h];
    atomicAdd(out + (size_t)d * HID + f, hg[(size_t)s * HID + f] * alpha);
}

// relu(agg + bg), partial mean reduction into pooled[HID]
__global__ void gat_pool(const float* __restrict__ agg, const float* __restrict__ bg,
                         float* __restrict__ pooled, int Nn) {
    int f = threadIdx.x;                             // blockDim = HID
    int n0 = blockIdx.x * 128;
    int n1 = n0 + 128 < Nn ? n0 + 128 : Nn;
    float b = bg[f];
    float s = 0.f;
    for (int n = n0; n < n1; ++n) {
        float v = agg[(size_t)n * HID + f] + b;
        s += v > 0.f ? v : 0.f;
    }
    atomicAdd(pooled + f, s);
}

// tiny classifier MLP, single block
__global__ __launch_bounds__(128)
void final_mlp(const float* __restrict__ pooled,
               const float* __restrict__ Wc1, const float* __restrict__ bc1,
               const float* __restrict__ Wc2, const float* __restrict__ bc2,
               const float* __restrict__ Wc3, const float* __restrict__ bc3,
               float* __restrict__ out, float invN) {
    __shared__ float p[HID], z1[128], z2[64];
    int t = threadIdx.x;
    p[t]       = pooled[t] * invN;
    p[t + 128] = pooled[t + 128] * invN;
    __syncthreads();
    float a1 = bc1[t];
    for (int k = 0; k < HID; ++k) a1 += p[k] * Wc1[k * 128 + t];
    z1[t] = a1 > 0.f ? a1 : 0.f;
    __syncthreads();
    if (t < 64) {
        float a2 = bc2[t];
        for (int k = 0; k < 128; ++k) a2 += z1[k] * Wc2[k * 64 + t];
        z2[t] = a2 > 0.f ? a2 : 0.f;
    }
    __syncthreads();
    if (t < 8) {
        float a3 = bc3[t];
        for (int k = 0; k < 64; ++k) a3 += z2[k] * Wc3[k * 8 + t];
        out[t] = a3;
    }
}

// --------------------------------------------------------------- launch ----
extern "C" void kernel_launch(void* const* d_in, const int* in_sizes, int n_in,
                              void* d_out, int out_size, void* d_ws, size_t ws_size,
                              hipStream_t stream) {
    const float* x     = (const float*)d_in[0];
    const int*   ei    = (const int*)d_in[1];
    const float* W1    = (const float*)d_in[2];
    const float* b1    = (const float*)d_in[3];
    const float* W2    = (const float*)d_in[4];
    const float* b2    = (const float*)d_in[5];
    const float* W3    = (const float*)d_in[6];
    const float* b3    = (const float*)d_in[7];
    const float* Wg    = (const float*)d_in[8];
    const float* a_src = (const float*)d_in[9];
    const float* a_dst = (const float*)d_in[10];
    const float* bg    = (const float*)d_in[11];
    const float* Wc1   = (const float*)d_in[12];
    const float* bc1   = (const float*)d_in[13];
    const float* Wc2   = (const float*)d_in[14];
    const float* bc2   = (const float*)d_in[15];
    const float* Wc3   = (const float*)d_in[16];
    const float* bc3   = (const float*)d_in[17];
    float* out = (float*)d_out;

    const int Nn = in_sizes[0] / D_IN;   // 50000
    const int E  = in_sizes[1] / 2;      // 800000
    const int EN = E + Nn;

    // ---- workspace carving ----
    char* base = (char*)d_ws;
    size_t off = 0;
    auto carve = [&](size_t bytes) -> char* {
        char* p = base + off;
        off = (off + bytes + 255) & ~(size_t)255;
        return p;
    };
    __bf16* XBF  = (__bf16*)carve((size_t)Nn * D_IN * 2);
    float*  T    = (float*) carve((size_t)Nn * HID * 4);   // GEMM out / hg
    float*  AGG  = (float*) carve((size_t)Nn * HID * 4);   // aggregation target
    float*  H1F  = (float*) carve((size_t)Nn * HID * 4);   // residual
    __bf16* HBF  = (__bf16*)carve((size_t)Nn * HID * 2);   // bf16 activations
    float*  DEG  = (float*) carve((size_t)Nn * 4);
    float*  DINV = (float*) carve((size_t)Nn * 4);
    float*  ALS  = (float*) carve((size_t)Nn * HEADS * 4);
    float*  ALD  = (float*) carve((size_t)Nn * HEADS * 4);
    float*  MMAX = (float*) carve((size_t)Nn * HEADS * 4);
    float*  DEN  = (float*) carve((size_t)Nn * HEADS * 4);
    __bf16* W1S  = (__bf16*)carve((size_t)D_IN * HID * 2);
    __bf16* W2S  = (__bf16*)carve((size_t)HID * HID * 2);
    __bf16* W3S  = (__bf16*)carve((size_t)HID * (HID / 2) * 2);
    __bf16* WGS  = (__bf16*)carve((size_t)(HID / 2) * HID * 2);
    float*  POOL = (float*) carve((size_t)HID * 4);

    const int FB = 256;   // generic block size
    auto blocks = [&](long long n) { return (int)((n + FB - 1) / FB); };

    // ---- prep: bf16 X, swizzled weights, degrees ----
    convert_f32_to_bf16<<<2048, FB, 0, stream>>>(x, XBF, (long long)Nn * D_IN);
    swizzle_weight<<<blocks(D_IN * HID), FB, 0, stream>>>(W1, W1S, D_IN, HID);
    swizzle_weight<<<blocks(HID * HID), FB, 0, stream>>>(W2, W2S, HID, HID);
    swizzle_weight<<<blocks(HID * (HID / 2)), FB, 0, stream>>>(W3, W3S, HID, HID / 2);
    swizzle_weight<<<blocks((HID / 2) * HID), FB, 0, stream>>>(Wg, WGS, HID / 2, HID);
    fill_f32<<<blocks(Nn), FB, 0, stream>>>(DEG, 0.f, Nn);
    compute_deg<<<blocks(E), FB, 0, stream>>>(ei, DEG, E);
    compute_dinv<<<blocks(Nn), FB, 0, stream>>>(DEG, DINV, Nn);

    // ---- GCN layer 1: T = X@W1 ; AGG = norm-agg(T) ; h1 = relu(AGG+b1) ----
    gemm_bf16_wmma<D_IN, HID><<<Nn / 16, 2 * HID, 0, stream>>>(XBF, W1S, T);
    fill_f32<<<2048, FB, 0, stream>>>(AGG, 0.f, (long long)Nn * HID);
    gcn_aggregate<<<EN, HID, 0, stream>>>(T, DINV, ei, AGG, E);
    gcn_epilogue<<<2048, FB, 0, stream>>>(AGG, b1, nullptr, H1F, HBF, HID,
                                          (long long)Nn * HID);

    // ---- GCN layer 2 (+residual h1) ----
    gemm_bf16_wmma<HID, HID><<<Nn / 16, 2 * HID, 0, stream>>>(HBF, W2S, T);
    fill_f32<<<2048, FB, 0, stream>>>(AGG, 0.f, (long long)Nn * HID);
    gcn_aggregate<<<EN, HID, 0, stream>>>(T, DINV, ei, AGG, E);
    gcn_epilogue<<<2048, FB, 0, stream>>>(AGG, b2, H1F, nullptr, HBF, HID,
                                          (long long)Nn * HID);

    // ---- GCN layer 3 (HID -> HID/2) ----
    gemm_bf16_wmma<HID, HID / 2><<<Nn / 16, HID, 0, stream>>>(HBF, W3S, T);
    fill_f32<<<2048, FB, 0, stream>>>(AGG, 0.f, (long long)Nn * (HID / 2));
    gcn_aggregate<<<EN, HID / 2, 0, stream>>>(T, DINV, ei, AGG, E);
    gcn_epilogue<<<2048, FB, 0, stream>>>(AGG, b3, nullptr, nullptr, HBF, HID / 2,
                                          (long long)Nn * (HID / 2));

    // ---- GAT: hg = h3@Wg ; attention softmax over incoming edges ----
    gemm_bf16_wmma<HID / 2, HID><<<Nn / 16, 2 * HID, 0, stream>>>(HBF, WGS, T);
    gat_node_attn<<<blocks((long long)Nn * HEADS), FB, 0, stream>>>(T, a_src, a_dst,
                                                                    ALS, ALD, Nn);
    fill_f32<<<blocks((long long)Nn * HEADS), FB, 0, stream>>>(MMAX, -INFINITY,
                                                               (long long)Nn * HEADS);
    fill_f32<<<blocks((long long)Nn * HEADS), FB, 0, stream>>>(DEN, 0.f,
                                                               (long long)Nn * HEADS);
    fill_f32<<<2048, FB, 0, stream>>>(AGG, 0.f, (long long)Nn * HID);
    gat_edge_max<<<blocks((long long)EN * HEADS), FB, 0, stream>>>(ei, ALS, ALD, MMAX, E, EN);
    gat_edge_den<<<blocks((long long)EN * HEADS), FB, 0, stream>>>(ei, ALS, ALD, MMAX, DEN, E, EN);
    gat_edge_agg<<<EN, HID, 0, stream>>>(ei, T, ALS, ALD, MMAX, DEN, AGG, E);

    // ---- mean pool + classifier ----
    fill_f32<<<1, FB, 0, stream>>>(POOL, 0.f, HID);
    gat_pool<<<(Nn + 127) / 128, HID, 0, stream>>>(AGG, bg, POOL, Nn);
    final_mlp<<<1, 128, 0, stream>>>(POOL, Wc1, bc1, Wc2, bc2, Wc3, bc3, out,
                                     1.0f / (float)Nn);
}
